// RGLRU_44083544326438
// MI455X (gfx1250) — compile-verified, matrix operand order checked by reading, not run on hardware
//
#include <hip/hip_runtime.h>
#include <hip/hip_bf16.h>

// ---------------------------------------------------------------------------
// RG-LRU (Griffin) block for MI455X / gfx1250.
// wave32 + WMMA bf16x3 GEMMs + async Global->LDS staging (ASYNCcnt path).
//   B=4, S=4096, D_IN=1024, D_HID=2048, D_INNER=1024, NB=8
// ---------------------------------------------------------------------------

typedef __attribute__((ext_vector_type(16))) __bf16 v16bf;
typedef __attribute__((ext_vector_type(8)))  float  v8f;

#define B_DIM   4
#define S_LEN   4096
#define D_IN    1024
#define D_HID   2048
#define D_INNER 1024
#define NTOK    (B_DIM * S_LEN)              // 16384 tokens
#define Y_ELEMS ((size_t)NTOK * D_IN)        // 16777216

#define WIN_N    (D_IN * D_HID)              // 2097152
#define WOUT_N   (D_HID * D_IN)              // 2097152
#define G_N      (8 * 128 * 256)             // 262144
#define XSPLIT_N ((size_t)NTOK * D_IN)       // 16777216

// ---------------------------------------------------------------------------
// CDNA5 helpers
// ---------------------------------------------------------------------------
__device__ __forceinline__ v8f v8f_zero() {
  v8f z = {0.f, 0.f, 0.f, 0.f, 0.f, 0.f, 0.f, 0.f};
  return z;
}

// Async Global -> LDS copy of 16 bytes (gfx1250, tracked by ASYNCcnt).
// lds_off: byte offset within the workgroup's LDS allocation (our single
// __shared__ array sits at offset 0). gaddr: 16B-aligned global address.
__device__ __forceinline__ void async_ld_b128(unsigned lds_off, const void* gaddr) {
  asm volatile("global_load_async_to_lds_b128 %0, %1, off"
               :: "v"(lds_off), "v"(gaddr)
               : "memory");
}

__device__ __forceinline__ void wait_async0() {
  asm volatile("s_wait_asynccnt 0x0" ::: "memory");
}

// Load a 16x32 bf16 fragment from a row-major tile (row stride `stride`).
// A-matrix layout (ISA §7.12.2): lanes 0-15 -> row M=lane, K {0..7,16..23};
// lanes 16-31 -> row M=lane-16, K {8..15,24..31}; 2 bf16 packed per VGPR.
// Also loads B (KxN) fragments when the source is stored transposed
// (N-major, K contiguous).
__device__ __forceinline__ v16bf frag_rm(const __bf16* base, int stride, int lane) {
  const int m  = lane & 15;
  const int kb = (lane & 16) ? 8 : 0;
  const __bf16* p = base + (size_t)m * stride + kb;
  v16bf r;
#pragma unroll
  for (int j = 0; j < 8; ++j) {
    r[j]     = p[j];        // K = kb + j
    r[8 + j] = p[16 + j];   // K = kb + 16 + j
  }
  return r;
}

__device__ __forceinline__ v8f wmma_bf16(v16bf a, v16bf b, v8f c) {
  // D = A(16x32) * B(32x16) + C, f32 accumulate
  return __builtin_amdgcn_wmma_f32_16x16x32_bf16(
      /*neg_a=*/false, a, /*neg_b=*/false, b,
      /*c_mod=*/(short)0, c, /*reuse_a=*/false, /*reuse_b=*/false);
}

// ---------------------------------------------------------------------------
// Kernel 0: prep — bf16 hi/lo split + transpose of weights (B operands get
// contiguous K runs) and bf16 hi/lo split of x (A operand planes, so the GEMM
// kernel can async-copy tiles to LDS without conversion).
// ---------------------------------------------------------------------------
__global__ __launch_bounds__(256) void k_prep(
    const float* __restrict__ w_in,  const float* __restrict__ w_out,
    const float* __restrict__ w_gx,  const float* __restrict__ w_ga,
    const float* __restrict__ x,
    __bf16* __restrict__ winT_hi,  __bf16* __restrict__ winT_lo,
    __bf16* __restrict__ woutT_hi, __bf16* __restrict__ woutT_lo,
    __bf16* __restrict__ gxT,      __bf16* __restrict__ gaT,
    __bf16* __restrict__ Xhi,      __bf16* __restrict__ Xlo)
{
  size_t idx = (size_t)blockIdx.x * 256 + threadIdx.x;
  if (idx < WIN_N) {                       // w_in (D_IN x D_HID) -> [n][k]
    int n = (int)(idx / D_IN), k = (int)(idx % D_IN);
    float v = w_in[(size_t)k * D_HID + n];
    __bf16 h = (__bf16)v;
    winT_hi[idx] = h;
    winT_lo[idx] = (__bf16)(v - (float)h);
    return;
  }
  idx -= WIN_N;
  if (idx < WOUT_N) {                      // w_out (D_HID x D_IN) -> [n][k]
    int n = (int)(idx / D_HID), k = (int)(idx % D_HID);
    float v = w_out[(size_t)k * D_IN + n];
    __bf16 h = (__bf16)v;
    woutT_hi[idx] = h;
    woutT_lo[idx] = (__bf16)(v - (float)h);
    return;
  }
  idx -= WOUT_N;
  if (idx < G_N) {                         // w_gx (8,256,128) -> [blk][o][i]
    int blk = (int)(idx / (128 * 256)), rem = (int)(idx % (128 * 256));
    int o = rem / 256, i = rem % 256;
    gxT[idx] = (__bf16)w_gx[((size_t)blk * 256 + i) * 128 + o];
    return;
  }
  idx -= G_N;
  if (idx < G_N) {
    int blk = (int)(idx / (128 * 256)), rem = (int)(idx % (128 * 256));
    int o = rem / 256, i = rem % 256;
    gaT[idx] = (__bf16)w_ga[((size_t)blk * 256 + i) * 128 + o];
    return;
  }
  idx -= G_N;
  if (idx < XSPLIT_N) {                    // x -> bf16 hi/lo planes (no reorder)
    float v = x[idx];
    __bf16 h = (__bf16)v;
    Xhi[idx] = h;
    Xlo[idx] = (__bf16)(v - (float)h);
  }
}

// ---------------------------------------------------------------------------
// Kernel 1: fused  xh = x@w_in + b_in  ->  gates (block-diag GEMMs, sigmoid)
//           -> complex decay a, gated/normalized input xc.
// One workgroup = 16 tokens, 256 threads = 8 waves.
// Wave w owns xh columns [256w,256w+256) and gate block w (cols [128w,128w+128)).
// LDS: one 64KB buffer, aliased: async-staged x hi/lo (16x256 each) during the
// big GEMM, then xh bf16 (16x2048) for gate GEMMs / elementwise stage.
// ---------------------------------------------------------------------------
__global__ __launch_bounds__(256) void k_fused_in(
    const __bf16* __restrict__ Xhi, const __bf16* __restrict__ Xlo,
    const float* __restrict__ b_in,
    const __bf16* __restrict__ winT_hi, const __bf16* __restrict__ winT_lo,
    const __bf16* __restrict__ gxT,     const __bf16* __restrict__ gaT,
    const float* __restrict__ b_gx, const float* __restrict__ b_ga,
    const float* __restrict__ arp,  const float* __restrict__ aip,
    float* __restrict__ Are, float* __restrict__ Aim,
    float* __restrict__ Xre, float* __restrict__ Xim)
{
  __shared__ __align__(16) __bf16 lds[16 * D_HID];   // 64 KB, LDS offset 0
  __bf16* xs_hi = lds;                 // [16][256]  (staging phase, off 0)
  __bf16* xs_lo = lds + 16 * 256;      // [16][256]  (off 8192 bytes)
  __bf16* xh    = lds;                 // [16][2048] (after big GEMM)

  const int tid  = threadIdx.x;
  const int wave = tid >> 5;
  const int lane = tid & 31;
  const int t0   = blockIdx.x * 16;    // token tile base (flat b*S+s)

  v8f acc[16];
#pragma unroll
  for (int i = 0; i < 16; ++i) acc[i] = v8f_zero();

  // ---- xh = x @ w_in : K=1024 in 4 chunks of 256, bf16x3 ----
  for (int kc = 0; kc < D_IN; kc += 256) {
    // async-stage the 16x256 hi/lo A-tiles (16B segments, 512 per plane)
    for (int s = tid; s < 512; s += 256) {
      int m  = s >> 5;                 // row 0..15
      int cb = (s & 31) * 16;          // byte offset within 512B row
      unsigned loff = (unsigned)(m * 512 + cb);
      size_t gb = ((size_t)(t0 + m) * D_IN + kc) * 2 + cb;
      async_ld_b128(loff,        (const char*)Xhi + gb);
      async_ld_b128(loff + 8192, (const char*)Xlo + gb);
    }
    wait_async0();
    __syncthreads();
#pragma unroll
    for (int nt = 0; nt < 16; ++nt) {
      const size_t bb = ((size_t)(wave * 256 + nt * 16)) * D_IN + kc;
      for (int k0 = 0; k0 < 256; k0 += 32) {
        v16bf ah = frag_rm(xs_hi + k0, 256, lane);
        v16bf al = frag_rm(xs_lo + k0, 256, lane);
        v16bf bh = frag_rm(winT_hi + bb + k0, D_IN, lane);
        v16bf bl = frag_rm(winT_lo + bb + k0, D_IN, lane);
        acc[nt] = wmma_bf16(ah, bh, acc[nt]);
        acc[nt] = wmma_bf16(ah, bl, acc[nt]);
        acc[nt] = wmma_bf16(al, bh, acc[nt]);
      }
    }
    __syncthreads();   // before restaging
  }

  // ---- write xh (+bias) as bf16 into LDS (C/D layout -> row-major tile) ----
  const int n  = lane & 15;
  const int m0 = (lane & 16) ? 8 : 0;
#pragma unroll
  for (int nt = 0; nt < 16; ++nt) {
    int c = wave * 256 + nt * 16 + n;
    float bias = b_in[c];
#pragma unroll
    for (int i = 0; i < 8; ++i)
      xh[(m0 + i) * D_HID + c] = (__bf16)(acc[nt][i] + bias);
  }
  __syncthreads();

  // ---- block-diagonal gate GEMMs: block = wave, K=256, plain bf16 ----
  v8f gx8[8], ga8[8];
#pragma unroll
  for (int g = 0; g < 8; ++g) { gx8[g] = v8f_zero(); ga8[g] = v8f_zero(); }
#pragma unroll
  for (int g = 0; g < 8; ++g) {
    const size_t gb = ((size_t)(wave * 128 + g * 16)) * 256;
    for (int k0 = 0; k0 < 256; k0 += 32) {
      v16bf a  = frag_rm(xh + wave * 256 + k0, D_HID, lane);
      v16bf bx = frag_rm(gxT + gb + k0, 256, lane);
      v16bf ba = frag_rm(gaT + gb + k0, 256, lane);
      gx8[g] = wmma_bf16(a, bx, gx8[g]);
      ga8[g] = wmma_bf16(a, ba, ga8[g]);
    }
  }

  // ---- elementwise: a = exp(-8*ga*softplus(arp) + i*aip*ga), xc = norm*gx*xh ----
#pragma unroll
  for (int g = 0; g < 8; ++g) {
    int c = wave * 128 + g * 16 + n;           // channel in [0,1024)
    float bgx  = b_gx[c];
    float bga  = b_ga[c];
    float sp   = logf(1.0f + expf(arp[c]));    // softplus(a_real_param)
    float ai_p = aip[c];
#pragma unroll
    for (int i = 0; i < 8; ++i) {
      int m = m0 + i;
      float gx = 1.0f / (1.0f + expf(-(gx8[g][i] + bgx)));
      float ga = 1.0f / (1.0f + expf(-(ga8[g][i] + bga)));
      float log_a = -8.0f * ga * sp;           // LOG_A_SCALE = -8
      float th    = ai_p * ga;
      float mag   = expf(log_a);
      float norm  = sqrtf(fmaxf(1.0f - mag * mag, 1e-6f));  // exp(2*log_a)=mag^2
      float scale = norm * gx;
      float xr = (float)xh[m * D_HID + c];
      float xi = (float)xh[m * D_HID + D_INNER + c];
      size_t o = (size_t)(t0 + m) * D_INNER + c;   // channel-contiguous
      Are[o] = mag * cosf(th);
      Aim[o] = mag * sinf(th);
      Xre[o] = scale * xr;
      Xim[o] = scale * xi;
    }
  }
}

// ---------------------------------------------------------------------------
// Kernel 2: complex linear recurrence h_t = a_t*h_{t-1} + xc_t over S.
// One thread per (b, channel): 4096 threads, coalesced per-step loads,
// gfx1250 global_prefetch lookahead. Emits h_m as bf16 hi/lo planes (ready
// for the async-staged bf16x3 output GEMM) and h_last in fp32.
// ---------------------------------------------------------------------------
__global__ __launch_bounds__(256) void k_scan(
    const float* __restrict__ Are, const float* __restrict__ Aim,
    const float* __restrict__ Xre, const float* __restrict__ Xim,
    __bf16* __restrict__ Hm_hi, __bf16* __restrict__ Hm_lo,
    float* __restrict__ hlast)
{
  int gid = blockIdx.x * 256 + threadIdx.x;    // 0..4095
  int b = gid >> 10;
  int c = gid & 1023;
  float hr = 0.f, hi = 0.f;
  size_t base = (size_t)b * S_LEN * D_INNER + c;
  size_t hb   = (size_t)b * S_LEN * D_HID  + c;
  for (int s = 0; s < S_LEN; ++s) {
    size_t o = base + (size_t)s * D_INNER;
    __builtin_prefetch(&Are[o + 16 * D_INNER], 0, 0);  // global_prefetch_b8
    __builtin_prefetch(&Xre[o + 16 * D_INNER], 0, 0);
    float ar = Are[o], ai = Aim[o], xr = Xre[o], xi = Xim[o];
    float nr = __builtin_fmaf(ar, hr, __builtin_fmaf(-ai, hi, xr));
    float ni = __builtin_fmaf(ar, hi, __builtin_fmaf( ai, hr, xi));
    hr = nr; hi = ni;
    __bf16 rh = (__bf16)hr;
    __bf16 ih = (__bf16)hi;
    size_t ho = hb + (size_t)s * D_HID;
    Hm_hi[ho]           = rh;
    Hm_lo[ho]           = (__bf16)(hr - (float)rh);
    Hm_hi[ho + D_INNER] = ih;
    Hm_lo[ho + D_INNER] = (__bf16)(hi - (float)ih);
  }
  hlast[b * D_HID + c]           = hr;
  hlast[b * D_HID + c + D_INNER] = hi;
}

// ---------------------------------------------------------------------------
// Kernel 3: y = h_m @ w_out + b_out  (16384x2048 @ 2048x1024), bf16x3 WMMA.
// 16 tokens per workgroup, 8 waves x 8 N-tiles; K async-staged in 256-chunks.
// ---------------------------------------------------------------------------
__global__ __launch_bounds__(256) void k_out_gemm(
    const __bf16* __restrict__ Hm_hi, const __bf16* __restrict__ Hm_lo,
    const __bf16* __restrict__ woutT_hi, const __bf16* __restrict__ woutT_lo,
    const float* __restrict__ b_out, float* __restrict__ y)
{
  __shared__ __align__(16) __bf16 hstage[2 * 16 * 256];  // 16 KB, LDS offset 0
  __bf16* hhi = hstage;               // off 0
  __bf16* hlo = hstage + 16 * 256;    // off 8192 bytes

  const int tid  = threadIdx.x;
  const int wave = tid >> 5;
  const int lane = tid & 31;
  const int t0   = blockIdx.x * 16;

  v8f acc[8];
#pragma unroll
  for (int i = 0; i < 8; ++i) acc[i] = v8f_zero();

  for (int kc = 0; kc < D_HID; kc += 256) {
    for (int s = tid; s < 512; s += 256) {
      int m  = s >> 5;
      int cb = (s & 31) * 16;
      unsigned loff = (unsigned)(m * 512 + cb);
      size_t gb = ((size_t)(t0 + m) * D_HID + kc) * 2 + cb;
      async_ld_b128(loff,        (const char*)Hm_hi + gb);
      async_ld_b128(loff + 8192, (const char*)Hm_lo + gb);
    }
    wait_async0();
    __syncthreads();
#pragma unroll
    for (int nt = 0; nt < 8; ++nt) {
      const size_t bb = ((size_t)(wave * 128 + nt * 16)) * D_HID + kc;
      for (int k0 = 0; k0 < 256; k0 += 32) {
        v16bf ah = frag_rm(hhi + k0, 256, lane);
        v16bf al = frag_rm(hlo + k0, 256, lane);
        v16bf bh = frag_rm(woutT_hi + bb + k0, D_HID, lane);
        v16bf bl = frag_rm(woutT_lo + bb + k0, D_HID, lane);
        acc[nt] = wmma_bf16(ah, bh, acc[nt]);
        acc[nt] = wmma_bf16(ah, bl, acc[nt]);
        acc[nt] = wmma_bf16(al, bh, acc[nt]);
      }
    }
    __syncthreads();
  }

  const int n  = lane & 15;
  const int m0 = (lane & 16) ? 8 : 0;
#pragma unroll
  for (int nt = 0; nt < 8; ++nt) {
    int c = wave * 128 + nt * 16 + n;
    float bias = b_out[c];
#pragma unroll
    for (int i = 0; i < 8; ++i)
      y[(size_t)(t0 + m0 + i) * D_IN + c] = acc[nt][i] + bias;
  }
}

// ---------------------------------------------------------------------------
// Launch: prep -> fused input -> scan -> output GEMM (all on `stream`).
// Workspace layout (~466 MB of d_ws).
// ---------------------------------------------------------------------------
extern "C" void kernel_launch(void* const* d_in, const int* in_sizes, int n_in,
                              void* d_out, int out_size, void* d_ws, size_t ws_size,
                              hipStream_t stream) {
  (void)in_sizes; (void)n_in; (void)out_size; (void)ws_size;

  const float* x     = (const float*)d_in[0];
  const float* w_in  = (const float*)d_in[1];
  const float* b_in  = (const float*)d_in[2];
  const float* w_gx  = (const float*)d_in[3];
  const float* b_gx  = (const float*)d_in[4];
  const float* w_ga  = (const float*)d_in[5];
  const float* b_ga  = (const float*)d_in[6];
  const float* arp   = (const float*)d_in[7];
  const float* aip   = (const float*)d_in[8];
  const float* w_out = (const float*)d_in[9];
  const float* b_out = (const float*)d_in[10];

  char* ws = (char*)d_ws;
  size_t off = 0;
  __bf16* winT_hi  = (__bf16*)(ws + off); off += (size_t)WIN_N  * 2;
  __bf16* winT_lo  = (__bf16*)(ws + off); off += (size_t)WIN_N  * 2;
  __bf16* woutT_hi = (__bf16*)(ws + off); off += (size_t)WOUT_N * 2;
  __bf16* woutT_lo = (__bf16*)(ws + off); off += (size_t)WOUT_N * 2;
  __bf16* gxT      = (__bf16*)(ws + off); off += (size_t)G_N * 2;
  __bf16* gaT      = (__bf16*)(ws + off); off += (size_t)G_N * 2;
  __bf16* Xhi      = (__bf16*)(ws + off); off += XSPLIT_N * 2;
  __bf16* Xlo      = (__bf16*)(ws + off); off += XSPLIT_N * 2;
  float*  Are      = (float*)(ws + off);  off += (size_t)NTOK * D_INNER * 4;
  float*  Aim      = (float*)(ws + off);  off += (size_t)NTOK * D_INNER * 4;
  float*  Xre      = (float*)(ws + off);  off += (size_t)NTOK * D_INNER * 4;
  float*  Xim      = (float*)(ws + off);  off += (size_t)NTOK * D_INNER * 4;
  __bf16* Hm_hi    = (__bf16*)(ws + off); off += (size_t)NTOK * D_HID * 2;
  __bf16* Hm_lo    = (__bf16*)(ws + off); off += (size_t)NTOK * D_HID * 2;

  float* y     = (float*)d_out;
  float* hlast = y + Y_ELEMS;   // (B, 2048) = h_m[:, -1]

  const size_t prep_total = (size_t)WIN_N + WOUT_N + 2 * G_N + XSPLIT_N;
  k_prep<<<(unsigned)((prep_total + 255) / 256), 256, 0, stream>>>(
      w_in, w_out, w_gx, w_ga, x,
      winT_hi, winT_lo, woutT_hi, woutT_lo, gxT, gaT, Xhi, Xlo);

  k_fused_in<<<NTOK / 16, 256, 0, stream>>>(
      Xhi, Xlo, b_in, winT_hi, winT_lo, gxT, gaT, b_gx, b_ga, arp, aip,
      Are, Aim, Xre, Xim);

  k_scan<<<(B_DIM * D_INNER) / 256, 256, 0, stream>>>(
      Are, Aim, Xre, Xim, Hm_hi, Hm_lo, hlast);

  k_out_gemm<<<NTOK / 16, 256, 0, stream>>>(
      Hm_hi, Hm_lo, woutT_hi, woutT_lo, b_out, y);
}
